// PostProcess_34969623724347
// MI455X (gfx1250) — 8-bit matrix operand order repaired and checked by reading, compiled, not run on hardware
//
#include <hip/hip_runtime.h>
#include <math.h>

typedef __attribute__((ext_vector_type(8))) int v8i;

#define NB   4096
#define NCH  85
#define NCLS 80
#define IOU_THR 0.5f

#define OFF_BOXES    0u
#define OFF_SCORES   65536u
#define OFF_CLS      81920u
#define OFF_SBOXES   98304u
#define OFF_SSCORES  163840u
#define OFF_SCLS     180224u
#define OFF_SUP      196608u
#define OFF_ALIVE    2293760u

__global__ __launch_bounds__(128) void decode_kernel(const float* __restrict__ pred,
                                                     float* __restrict__ boxes,
                                                     float* __restrict__ scores,
                                                     int* __restrict__ cls) {
  const int wave = threadIdx.x >> 5;
  const int lane = threadIdx.x & 31;
  const int box  = blockIdx.x * 4 + wave;
  if (box >= NB) return;
  const float* p = pred + box * NCH;

  float l0 = (lane      < NCLS) ? p[5 + lane]      : -1e30f;
  float l1 = (lane + 32 < NCLS) ? p[5 + lane + 32] : -1e30f;
  float l2 = (lane + 64 < NCLS) ? p[5 + lane + 64] : -1e30f;

  float m = l0; int a = lane;
  if (l1 > m) { m = l1; a = lane + 32; }
  if (l2 > m) { m = l2; a = lane + 64; }
  #pragma unroll
  for (int off = 16; off > 0; off >>= 1) {
    float om = __shfl_xor(m, off, 32);
    int   oa = __shfl_xor(a, off, 32);
    if (om > m || (om == m && oa < a)) { m = om; a = oa; }
  }
  float s = expf(l0 - m) + expf(l1 - m) + expf(l2 - m);
  #pragma unroll
  for (int off = 16; off > 0; off >>= 1) s += __shfl_xor(s, off, 32);

  if (lane == 0) {
    float x = p[0], y = p[1], w = p[2], h = p[3], cf = p[4];
    boxes[box * 4 + 0] = x - 0.5f * w;
    boxes[box * 4 + 1] = y - 0.5f * h;
    boxes[box * 4 + 2] = x + 0.5f * w;
    boxes[box * 4 + 3] = y + 0.5f * h;
    float conf = 1.0f / (1.0f + expf(-cf));
    scores[box] = conf * (1.0f / s);
    cls[box] = a;
  }
}

__global__ __launch_bounds__(256) void rank_kernel(const float* __restrict__ boxes,
                                                   const float* __restrict__ scores,
                                                   const int* __restrict__ cls,
                                                   float* __restrict__ sboxes,
                                                   float* __restrict__ sscores,
                                                   int* __restrict__ scls) {
  const int i = blockIdx.x * blockDim.x + threadIdx.x;
  if (i >= NB) return;
  const float si = scores[i];
  int r = 0;
  for (int j = 0; j < NB; ++j) {
    const float sj = scores[j];
    r += (sj > si) || (sj == si && j < i);
  }
  sboxes[r * 4 + 0] = boxes[i * 4 + 0];
  sboxes[r * 4 + 1] = boxes[i * 4 + 1];
  sboxes[r * 4 + 2] = boxes[i * 4 + 2];
  sboxes[r * 4 + 3] = boxes[i * 4 + 3];
  sscores[r] = si;
  scls[r] = cls[i];
}

__global__ __launch_bounds__(128) void sup_kernel(const float* __restrict__ sboxes,
                                                  const int* __restrict__ scls,
                                                  unsigned short* __restrict__ sup) {
  const int lane    = threadIdx.x & 31;
  const int strip   = blockIdx.x * 4 + (threadIdx.x >> 5);
  const int tileRow = blockIdx.y;
  const int l15 = lane & 15;
  const int hi  = (lane >> 4) & 1;
  const float4* __restrict__ sb4 = (const float4*)sboxes;

  const int clsRow = scls[tileRow * 16 + l15];
  v8i A0, A1;
  #pragma unroll
  for (int v = 0; v < 8; ++v) {
    int a0 = 0, a1 = 0;
    #pragma unroll
    for (int b = 0; b < 4; ++b) {
      const int ka = (v >> 1) * 16 + (v & 1) * 4 + b + hi * 8;
      a0 |= ((clsRow == ka)      ? 1 : 0) << (8 * b);
      a1 |= ((clsRow == ka + 64) ? 1 : 0) << (8 * b);
    }
    A0[v] = a0; A1[v] = a1;
  }

  float ix1[8], iy1[8], ix2[8], iy2[8], areaI[8];
  #pragma unroll
  for (int r = 0; r < 8; ++r) {
    const float4 bi = sb4[tileRow * 16 + r + hi * 8];
    ix1[r] = bi.x; iy1[r] = bi.y; ix2[r] = bi.z; iy2[r] = bi.w;
    areaI[r] = fmaxf(bi.z - bi.x, 0.f) * fmaxf(bi.w - bi.y, 0.f);
  }

  #pragma unroll
  for (int t = 0; t < 4; ++t) {
    const int tileCol = strip * 4 + t;

    const int clsCol = scls[tileCol * 16 + l15];
    v8i B0, B1;
    #pragma unroll
    for (int v = 0; v < 8; ++v) {
      int b0 = 0, b1 = 0;
      #pragma unroll
      for (int b = 0; b < 4; ++b) {
        const int kb = (v >> 1) * 16 + (v & 1) * 4 + b + hi * 8;
        b0 |= ((clsCol == kb)      ? 1 : 0) << (8 * b);
        b1 |= ((clsCol == kb + 64) ? 1 : 0) << (8 * b);
      }
      B0[v] = b0; B1[v] = b1;
    }

    v8i C = {};
    C = __builtin_amdgcn_wmma_i32_16x16x64_iu8(false, A0, false, B0, C, false, false);
    C = __builtin_amdgcn_wmma_i32_16x16x64_iu8(false, A1, false, B1, C, false, false);

    const float4 bj = sb4[tileCol * 16 + l15];
    const float areaJ = fmaxf(bj.z - bj.x, 0.f) * fmaxf(bj.w - bj.y, 0.f);

    #pragma unroll
    for (int r = 0; r < 8; ++r) {
      const float iw = fmaxf(fminf(ix2[r], bj.z) - fmaxf(ix1[r], bj.x), 0.f);
      const float ih = fmaxf(fminf(iy2[r], bj.w) - fmaxf(iy1[r], bj.y), 0.f);
      const float inter = iw * ih;
      const bool sbit = (inter > IOU_THR * (areaI[r] + areaJ - inter + 1e-9f))
                        && (C[r] != 0);
      const unsigned mask = (unsigned)__builtin_amdgcn_ballot_w32(sbit);
      if (lane == 0) {
        sup[(tileRow * 16 + r)     * 256 + tileCol] = (unsigned short)(mask & 0xFFFFu);
        sup[(tileRow * 16 + r + 8) * 256 + tileCol] = (unsigned short)(mask >> 16);
      }
    }
  }
}

__global__ __launch_bounds__(32) void nms_kernel(const unsigned* __restrict__ sup32,
                                                 unsigned* __restrict__ aliveOut) {
  const int lane = threadIdx.x;
  unsigned alive[4] = {0xFFFFFFFFu, 0xFFFFFFFFu, 0xFFFFFFFFu, 0xFFFFFFFFu};

  #pragma unroll
  for (int seg = 0; seg < 4; ++seg) {
    for (int i = seg * 1024; i < (seg + 1) * 1024; ++i) {
      const int word = i >> 5, bit = i & 31;
      const unsigned ownerWord = __shfl(alive[seg], word & 31, 32);
      if ((ownerWord >> bit) & 1u) {
        #pragma unroll
        for (int s = 0; s < 4; ++s) {
          const int w = lane + 32 * s;
          const unsigned msup = sup32[i * 128 + w];
          unsigned gt;
          if (w > word)       gt = 0xFFFFFFFFu;
          else if (w == word) gt = ~(0xFFFFFFFFu >> (31 - bit));
          else                gt = 0u;
          alive[s] &= ~(msup & gt);
        }
      }
    }
  }
  #pragma unroll
  for (int s = 0; s < 4; ++s) aliveOut[lane + 32 * s] = alive[s];
}

__global__ __launch_bounds__(256) void emit_kernel(const float* __restrict__ sboxes,
                                                   const float* __restrict__ sscores,
                                                   const int* __restrict__ scls,
                                                   const unsigned* __restrict__ alive,
                                                   float* __restrict__ out) {
  const int i = blockIdx.x * blockDim.x + threadIdx.x;
  if (i >= NB) return;
  const float kf = ((alive[i >> 5] >> (i & 31)) & 1u) ? 1.0f : 0.0f;
  out[i * 4 + 0] = sboxes[i * 4 + 0] * kf;
  out[i * 4 + 1] = sboxes[i * 4 + 1] * kf;
  out[i * 4 + 2] = sboxes[i * 4 + 2] * kf;
  out[i * 4 + 3] = sboxes[i * 4 + 3] * kf;
  out[NB * 4 + i] = sscores[i] * kf;
  out[NB * 5 + i] = (float)scls[i];
  out[NB * 6 + i] = kf;
}

extern "C" void kernel_launch(void* const* d_in, const int* in_sizes, int n_in,
                              void* d_out, int out_size, void* d_ws, size_t ws_size,
                              hipStream_t stream) {
  (void)in_sizes; (void)n_in; (void)out_size; (void)ws_size;
  const float* pred = (const float*)d_in[0];
  char* ws = (char*)d_ws;

  float* boxes    = (float*)(ws + OFF_BOXES);
  float* scores   = (float*)(ws + OFF_SCORES);
  int*   cls      = (int*)  (ws + OFF_CLS);
  float* sboxes   = (float*)(ws + OFF_SBOXES);
  float* sscores  = (float*)(ws + OFF_SSCORES);
  int*   scls     = (int*)  (ws + OFF_SCLS);
  unsigned short* sup   = (unsigned short*)(ws + OFF_SUP);
  unsigned*       sup32 = (unsigned*)      (ws + OFF_SUP);
  unsigned*       aliv  = (unsigned*)      (ws + OFF_ALIVE);
  float* out = (float*)d_out;

  decode_kernel<<<NB / 4, 128, 0, stream>>>(pred, boxes, scores, cls);
  rank_kernel<<<NB / 256, 256, 0, stream>>>(boxes, scores, cls, sboxes, sscores, scls);
  sup_kernel<<<dim3(16, 256), 128, 0, stream>>>(sboxes, scls, sup);
  nms_kernel<<<1, 32, 0, stream>>>(sup32, aliv);
  emit_kernel<<<NB / 256, 256, 0, stream>>>(sboxes, sscores, scls, aliv, out);
}
